// DotAttention_30966714204688
// MI455X (gfx1250) — matrix operand order repaired and checked by reading, not compiled, not run on hardware
//
#include <hip/hip_runtime.h>
#include <hip/hip_bf16.h>

typedef __attribute__((ext_vector_type(16))) __bf16 bf16x16;
typedef __attribute__((ext_vector_type(8)))  __bf16 bf16x8;
typedef __attribute__((ext_vector_type(4)))  __bf16 bf16x4;
typedef __attribute__((ext_vector_type(8)))  float  f32x8;
typedef __attribute__((ext_vector_type(4)))  float  f32x4;
typedef __attribute__((ext_vector_type(4)))  unsigned int u32x4;
typedef __attribute__((ext_vector_type(8)))  unsigned int u32x8;

#define B_   32
#define TE_  1024
#define TD_  512
#define D_   512
#define NS_  512

__device__ __forceinline__ bf16x8 ld8(const __bf16* p) { return *(const bf16x8*)p; }
__device__ __forceinline__ bf16x16 cat16(bf16x8 a, bf16x8 b) {
  return __builtin_shufflevector(a, b, 0,1,2,3,4,5,6,7,8,9,10,11,12,13,14,15);
}
// A fragment (16x32 bf16, MxK): lane<16 -> K [0..16); lane>=16 -> K [16..32)
__device__ __forceinline__ bf16x16 loadA(const __bf16* row, int hi) {
  const __bf16* p = row + hi * 16;
  return cat16(ld8(p), ld8(p + 8));
}
// B fragment (32x16 bf16, KxN): lane<16 -> K [0..16); lane>=16 -> K [16..32)
__device__ __forceinline__ bf16x16 loadB(const __bf16* col, int hi) {
  const __bf16* p = col + hi * 16;
  return cat16(ld8(p), ld8(p + 8));
}
__device__ __forceinline__ f32x8 wmma_bf16(bf16x16 a, bf16x16 b, f32x8 c) {
  return __builtin_amdgcn_wmma_f32_16x16x32_bf16(false, a, false, b, (short)0, c,
                                                 false, false);
}

// ---- Tensor Data Mover: DMA a 2-D bf16 tile (tile1 rows x tile0 elems) into LDS.
// Descriptor per CDNA5 ISA 8.3/8.4. pad: +4 dwords (16B) after every 256 dwords
// (1024B) stored -> LDS row stride gets a 4-bank rotation (conflict avoidance).
__device__ __forceinline__ void tdm_load_2d_bf16(
    unsigned lds_addr, const void* gptr,
    unsigned dim0, unsigned dim1, unsigned stride0,
    unsigned tile0, unsigned tile1)
{
  unsigned long long ga = (unsigned long long)(uintptr_t)gptr;
  u32x4 g0;
  g0[0] = 1u;                                    // count=1 (valid), user-mode
  g0[1] = lds_addr;                              // lds_addr (bytes)
  g0[2] = (unsigned)ga;                          // global_addr[31:0]
  g0[3] = ((unsigned)(ga >> 32) & 0x01ffffffu)   // global_addr[56:32]
        | (2u << 30);                            // type = 2 ("image")
  u32x8 g1;
  g1[0] = (1u << 16)       // data_size = 1 -> 2-byte elements
        | (1u << 20)       // pad_enable
        | (7u << 22)       // pad_interval code 7 -> 256 dwords
        | (3u << 25);      // pad_amount  code 3 -> 4 dwords
  g1[1] = (dim0 & 0xffffu) << 16;                          // tensor_dim0[15:0]
  g1[2] = (dim0 >> 16) | ((dim1 & 0xffffu) << 16);         // dim0[31:16]|dim1[15:0]
  g1[3] = (dim1 >> 16) | ((tile0 & 0xffffu) << 16);        // dim1[31:16]|tile_dim0
  g1[4] = tile1 & 0xffffu;                                 // tile_dim1 (tile_dim2=0)
  g1[5] = stride0;                                         // tensor_dim0_stride[31:0]
  g1[6] = 0u;                                              // stride0[47:32]|stride1 lo
  g1[7] = 0u;
  asm volatile("tensor_load_to_lds %0, %1" :: "s"(g0), "s"(g1) : "memory");
}

// ---------------- conversion kernels ----------------
__global__ void cvt_f32_bf16(const float* __restrict__ in, __bf16* __restrict__ out,
                             int n4) {
  int i = blockIdx.x * blockDim.x + threadIdx.x;
  if (i < n4) {
    float4 f = ((const float4*)in)[i];
    bf16x4 o = { (__bf16)f.x, (__bf16)f.y, (__bf16)f.z, (__bf16)f.w };
    ((bf16x4*)out)[i] = o;
  }
}

// out[n][d] = (bf16) in[d][n]   (LDS tile transpose, rows = D_, cols = NS_)
__global__ void transpose_cvt(const float* __restrict__ in, __bf16* __restrict__ out,
                              int rows, int cols) {
  __shared__ float tile[32][33];
  int c0 = blockIdx.x * 32, r0 = blockIdx.y * 32;
  int tx = threadIdx.x & 31, ty = threadIdx.x >> 5;
  for (int rr = ty; rr < 32; rr += 8)
    tile[rr][tx] = in[(size_t)(r0 + rr) * cols + c0 + tx];
  __syncthreads();
  for (int rr = ty; rr < 32; rr += 8)
    out[(size_t)(c0 + rr) * rows + r0 + tx] = (__bf16)tile[tx][rr];
}

// ---------------- K/V projection:  K = enc@Wk+bk (row-major),  Vt = (enc@Wv+bv)^T ----
__global__ void __launch_bounds__(256) proj_kv(
    const __bf16* __restrict__ encB,   // [B][TE][D]
    const __bf16* __restrict__ WkT,    // [NS][D]
    const __bf16* __restrict__ WvT,    // [NS][D]
    const float*  __restrict__ bk,
    const float*  __restrict__ bv,
    __bf16* __restrict__ Kb,           // [B][TE][NS]  row-major
    __bf16* __restrict__ Vt)           // [B][NS][TE]  transposed
{
  const int wave = threadIdx.x >> 5, lane = threadIdx.x & 31;
  const int ln = lane & 15, hi = lane >> 4;
  const int b  = blockIdx.x >> 3;                       // TE_/128 == 8
  const int t0 = (blockIdx.x & 7) * 128 + wave * 16;
  const __bf16* encRow = encB + ((size_t)b * TE_ + t0 + ln) * D_;

  for (int nc = 0; nc < NS_; nc += 64) {
    f32x8 accK[4] = {};
    f32x8 accV[4] = {};
    for (int k0 = 0; k0 < D_; k0 += 32) {
      bf16x16 a = loadA(encRow + k0, hi);
      __builtin_prefetch(encRow + k0 + 32, 0, 0);
#pragma unroll
      for (int j = 0; j < 4; ++j) {
        const __bf16* colK = WkT + (size_t)(nc + j * 16 + ln) * D_ + k0;
        const __bf16* colV = WvT + (size_t)(nc + j * 16 + ln) * D_ + k0;
        accK[j] = wmma_bf16(a, loadB(colK, hi), accK[j]);
        accV[j] = wmma_bf16(a, loadB(colV, hi), accV[j]);
      }
    }
#pragma unroll
    for (int j = 0; j < 4; ++j) {
      const int n = nc + j * 16 + ln;
      const float biasK = bk[n], biasV = bv[n];
#pragma unroll
      for (int r = 0; r < 8; ++r) {
        const int t = t0 + 8 * hi + r;
        Kb[((size_t)b * TE_ + t) * NS_ + n] = (__bf16)(accK[j][r] + biasK);
      }
      bf16x8 vv;
#pragma unroll
      for (int r = 0; r < 8; ++r) vv[r] = (__bf16)(accV[j][r] + biasV);
      *(bf16x8*)(Vt + ((size_t)b * NS_ + n) * TE_ + t0 + 8 * hi) = vv;
    }
  }
}

// ---------------- score^T tiles + softmax over t_enc ----------------
// dec tile (16 q rows x 512) is shared by all 8 waves -> TDM DMA into LDS once.
// Padded row stride: 512 elems + 8 pad elems = 520 (1040B -> 4-bank rotation/row).
__global__ void __launch_bounds__(256) score_softmax(
    const __bf16* __restrict__ decB,   // [B][TD][NS]
    const __bf16* __restrict__ Kb,     // [B][TE][NS]
    float*  __restrict__ attnF,        // [B][TD][TE] (output)
    __bf16* __restrict__ attnB)        // [B][TD][TE] (workspace, bf16)
{
  __shared__ __align__(16) __bf16 decS[16 * 520];
  __shared__ float red[8][32];
  const int wave = threadIdx.x >> 5, lane = threadIdx.x & 31;
  const int ln = lane & 15, hi = lane >> 4;
  const int b  = blockIdx.x >> 5;                      // TD_/16 == 32
  const int q0 = (blockIdx.x & 31) * 16;
  const int tbase = wave * 128;

  if (threadIdx.x < 32) {                              // wave 0 issues the DMA
    tdm_load_2d_bf16((unsigned)(uintptr_t)(void*)decS,
                     decB + ((size_t)b * TD_ + q0) * NS_,
                     NS_, 16, NS_, NS_, 16);
    __builtin_amdgcn_s_wait_tensorcnt(0);
  }
  __syncthreads();

  const __bf16* decRow = decS + ln * 520;              // this lane's q row in LDS
  f32x8 acc[8] = {};
  for (int k0 = 0; k0 < NS_; k0 += 32) {
    bf16x16 bfrag = loadB(decRow + k0, hi);            // ds_load_b128 x2
#pragma unroll
    for (int j = 0; j < 8; ++j) {
      const __bf16* arow = Kb + ((size_t)b * TE_ + tbase + j * 16 + ln) * NS_ + k0;
      acc[j] = wmma_bf16(loadA(arow, hi), bfrag, acc[j]);
    }
  }
  // Each lane holds q = q0+ln fixed for all its 64 score values (over t_enc).
  float m = -1e30f;
#pragma unroll
  for (int j = 0; j < 8; ++j)
#pragma unroll
    for (int r = 0; r < 8; ++r) m = fmaxf(m, acc[j][r]);
  red[wave][lane] = m;
  __syncthreads();
  float M = -1e30f;
#pragma unroll
  for (int w2 = 0; w2 < 8; ++w2)
    M = fmaxf(M, fmaxf(red[w2][ln], red[w2][16 + ln]));
  __syncthreads();
  float s = 0.f;
#pragma unroll
  for (int j = 0; j < 8; ++j)
#pragma unroll
    for (int r = 0; r < 8; ++r) {
      float e = __expf(acc[j][r] - M);
      acc[j][r] = e;
      s += e;
    }
  red[wave][lane] = s;
  __syncthreads();
  float S = 0.f;
#pragma unroll
  for (int w2 = 0; w2 < 8; ++w2) S += red[w2][ln] + red[w2][16 + ln];
  const float inv = 1.0f / S;

#pragma unroll
  for (int j = 0; j < 8; ++j) {
    const int t = tbase + j * 16 + 8 * hi;             // M = 8*hi + r
    float* op = attnF + ((size_t)b * TD_ + q0 + ln) * TE_ + t;
    f32x4 lo = { acc[j][0] * inv, acc[j][1] * inv, acc[j][2] * inv, acc[j][3] * inv };
    f32x4 hi4 = { acc[j][4] * inv, acc[j][5] * inv, acc[j][6] * inv, acc[j][7] * inv };
    *(f32x4*)op = lo;
    *(f32x4*)(op + 4) = hi4;
    bf16x8 bb;
#pragma unroll
    for (int r = 0; r < 8; ++r) bb[r] = (__bf16)(acc[j][r] * inv);
    *(bf16x8*)(attnB + ((size_t)b * TD_ + q0 + ln) * TE_ + t) = bb;
  }
}

// ---------------- context^T tiles: context = attn @ V ----------------
// attn tile (16 q rows x 1024) shared by all waves -> TDM into LDS once.
// Padded stride: 1024 elems + 2x8 pad = 1040 elems (2080B). Element k lives at
// byte k*2 + 16*(k>=512) within a row (pad after each 1024B segment).
__global__ void __launch_bounds__(256) context_gemm(
    const __bf16* __restrict__ attnB,  // [B][TD][TE]
    const __bf16* __restrict__ Vt,     // [B][NS][TE]
    float* __restrict__ ctx)           // [B][TD][NS]
{
  __shared__ __align__(16) __bf16 attnS[16 * 1040];
  const int wave = threadIdx.x >> 5, lane = threadIdx.x & 31;
  const int ln = lane & 15, hi = lane >> 4;
  const int b  = blockIdx.x >> 5;                      // TD_/16 == 32
  const int q0 = (blockIdx.x & 31) * 16;
  const int nbase = wave * 64;

  if (threadIdx.x < 32) {
    tdm_load_2d_bf16((unsigned)(uintptr_t)(void*)attnS,
                     attnB + ((size_t)b * TD_ + q0) * TE_,
                     TE_, 16, TE_, TE_, 16);
    __builtin_amdgcn_s_wait_tensorcnt(0);
  }
  __syncthreads();

  const __bf16* attnRow = attnS + ln * 1040;
  f32x8 acc[4] = {};
  for (int k0 = 0; k0 < TE_; k0 += 32) {
    const int ke = k0 + hi * 16;
    const __bf16* p = attnRow + ke + ((ke >> 9) << 3);   // skip mid-row pad
    bf16x16 bfrag = cat16(ld8(p), ld8(p + 8));           // ds_load_b128 x2
#pragma unroll
    for (int j = 0; j < 4; ++j) {
      const __bf16* arow = Vt + ((size_t)b * NS_ + nbase + j * 16 + ln) * TE_ + k0;
      __builtin_prefetch(arow + 32, 0, 0);
      acc[j] = wmma_bf16(loadA(arow, hi), bfrag, acc[j]);
    }
  }
#pragma unroll
  for (int j = 0; j < 4; ++j) {
    const int n = nbase + j * 16 + 8 * hi;             // M = 8*hi + r
    float* op = ctx + ((size_t)b * TD_ + q0 + ln) * NS_ + n;
    f32x4 lo = { acc[j][0], acc[j][1], acc[j][2], acc[j][3] };
    f32x4 hi4 = { acc[j][4], acc[j][5], acc[j][6], acc[j][7] };
    *(f32x4*)op = lo;
    *(f32x4*)(op + 4) = hi4;
  }
}

// ---------------- host side ----------------
extern "C" void kernel_launch(void* const* d_in, const int* in_sizes, int n_in,
                              void* d_out, int out_size, void* d_ws, size_t ws_size,
                              hipStream_t stream) {
  (void)in_sizes; (void)n_in; (void)out_size; (void)ws_size;
  const float* enc = (const float*)d_in[0];
  const float* dec = (const float*)d_in[1];
  const float* Wk  = (const float*)d_in[2];
  const float* bk  = (const float*)d_in[3];
  const float* Wv  = (const float*)d_in[4];
  const float* bv  = (const float*)d_in[5];

  float* ctx_out  = (float*)d_out;                          // [B][TD][NS]
  float* attn_out = ctx_out + (size_t)B_ * TD_ * NS_;       // [B][TD][TE]

  char* w = (char*)d_ws;
  __bf16* encB = (__bf16*)w;  w += (size_t)B_ * TE_ * D_  * 2;   // 32 MB
  __bf16* decB = (__bf16*)w;  w += (size_t)B_ * TD_ * NS_ * 2;   // 16 MB
  __bf16* WkT  = (__bf16*)w;  w += (size_t)NS_ * D_ * 2;         // 0.5 MB
  __bf16* WvT  = (__bf16*)w;  w += (size_t)NS_ * D_ * 2;         // 0.5 MB
  __bf16* Kb   = (__bf16*)w;  w += (size_t)B_ * TE_ * NS_ * 2;   // 32 MB
  __bf16* Vt   = (__bf16*)w;  w += (size_t)B_ * NS_ * TE_ * 2;   // 32 MB
  __bf16* attnBf = (__bf16*)w; w += (size_t)B_ * TD_ * TE_ * 2;  // 32 MB

  {
    int n4 = B_ * TE_ * D_ / 4;
    cvt_f32_bf16<<<n4 / 256, 256, 0, stream>>>(enc, encB, n4);
  }
  {
    int n4 = B_ * TD_ * NS_ / 4;
    cvt_f32_bf16<<<n4 / 256, 256, 0, stream>>>(dec, decB, n4);
  }
  {
    dim3 g(NS_ / 32, D_ / 32);
    transpose_cvt<<<g, 256, 0, stream>>>(Wk, WkT, D_, NS_);
    transpose_cvt<<<g, 256, 0, stream>>>(Wv, WvT, D_, NS_);
  }
  proj_kv<<<B_ * TE_ / 128, 256, 0, stream>>>(encB, WkT, WvT, bk, bv, Kb, Vt);
  score_softmax<<<B_ * (TD_ / 16), 256, 0, stream>>>(decB, Kb, attn_out, attnBf);
  context_gemm<<<B_ * (TD_ / 16), 256, 0, stream>>>(attnBf, Vt, ctx_out);
}
